// XTAttention_24395414242077
// MI455X (gfx1250) — matrix operand order repaired and checked by reading, not compiled, not run on hardware
//
#include <hip/hip_runtime.h>
#include <hip/hip_bf16.h>

typedef __attribute__((ext_vector_type(16))) _Float16 v16h;
typedef __attribute__((ext_vector_type(8)))  _Float16 v8h;
typedef __attribute__((ext_vector_type(8)))  float    v8f;
typedef __attribute__((ext_vector_type(4)))  float    v4f;

#define DIMX   1024
#define NSEQ   2048
#define NBAT   2
#define HEADS  16
#define HD     64
#define KDIM   192   // 3 concatenated streams of head_dim=64

// ---------------------------------------------------------------------------
// WMMA fragment loaders (gfx1250 wave32 layouts, CDNA5 ISA §7.12.2)
// A (16x32 f16): lane L -> row m=L&15, halves k=(e<8?0:16)+(L>>4)*8+(e&7)
// B (32x16 f16) from column-major source: lane L -> col n=L&15, k=(e<8?0:16)+(L>>4)*8+(e&7)
// C/D (16x16 f32): VGPR r -> [m=(L>>4)*8+r][n=L&15]
// ---------------------------------------------------------------------------
__device__ __forceinline__ v16h load_frag_a(const _Float16* A, int lda, int lane) {
    const _Float16* p = A + (size_t)(lane & 15) * lda + ((lane >> 4) << 3);
    v16h r;
    ((v8h*)&r)[0] = *(const v8h*)(p);
    ((v8h*)&r)[1] = *(const v8h*)(p + 16);
    return r;
}

__device__ __forceinline__ v16h load_frag_b(const _Float16* Bt, int ldb, int lane) {
    const _Float16* p = Bt + (size_t)(lane & 15) * ldb + ((lane >> 4) << 3);
    v16h r;
    ((v8h*)&r)[0] = *(const v8h*)(p);
    ((v8h*)&r)[1] = *(const v8h*)(p + 16);
    return r;
}

#define WMMA_F16(a, b, c) \
    __builtin_amdgcn_wmma_f32_16x16x32_f16(false, (a), false, (b), (short)0, (c), false, false)

// ---------------------------------------------------------------------------
// fp32 -> f16 conversion (8 elements / thread)
// ---------------------------------------------------------------------------
__global__ __launch_bounds__(256) void cvt_f32_f16(const float* __restrict__ src,
                                                   _Float16* __restrict__ dst, int n) {
    int i = (blockIdx.x * 256 + threadIdx.x) * 8;
    if (i + 8 <= n) {
        v4f a = *(const v4f*)(src + i);
        v4f b = *(const v4f*)(src + i + 4);
        v8h o;
        o[0] = (_Float16)a[0]; o[1] = (_Float16)a[1];
        o[2] = (_Float16)a[2]; o[3] = (_Float16)a[3];
        o[4] = (_Float16)b[0]; o[5] = (_Float16)b[1];
        o[6] = (_Float16)b[2]; o[7] = (_Float16)b[3];
        *(v8h*)(dst + i) = o;
    }
}

// ---------------------------------------------------------------------------
// Projections: 7 GEMMs Y = X @ W^T, software-pipelined (double-buffered frags
// so WMMAs overlap the next k-step's loads instead of s_wait_loadcnt 0).
// ---------------------------------------------------------------------------
__global__ __launch_bounds__(256) void proj_kernel(
    const _Float16* __restrict__ xw, const _Float16* __restrict__ xp,
    const _Float16* __restrict__ xc, const _Float16* __restrict__ wbase,
    _Float16* __restrict__ qb, _Float16* __restrict__ kb, _Float16* __restrict__ vb) {
    const int g    = blockIdx.y;
    const int lane = threadIdx.x & 31;
    const int wave = threadIdx.x >> 5;
    const int t    = blockIdx.x * 8 + wave;   // 0..4095 wave tiles
    const int m0   = (t >> 4) << 4;           // row tile (16)
    const int n0   = (t & 15) << 6;           // col tile (64)

    const int xi = (g < 3) ? g : (g < 6) ? (g - 3) : 0;
    const _Float16* xs = (xi == 0) ? xw : (xi == 1) ? xp : xc;
    const _Float16* W  = wbase + (size_t)g * (DIMX * DIMX);

    const _Float16* Ab = xs + (size_t)m0 * DIMX;
    const _Float16* W0 = W + (size_t)(n0 +  0) * DIMX;
    const _Float16* W1 = W + (size_t)(n0 + 16) * DIMX;
    const _Float16* W2 = W + (size_t)(n0 + 32) * DIMX;
    const _Float16* W3 = W + (size_t)(n0 + 48) * DIMX;

    v16h a_c  = load_frag_a(Ab, DIMX, lane);
    v16h b0_c = load_frag_b(W0, DIMX, lane);
    v16h b1_c = load_frag_b(W1, DIMX, lane);
    v16h b2_c = load_frag_b(W2, DIMX, lane);
    v16h b3_c = load_frag_b(W3, DIMX, lane);

    v8f acc0 = {}, acc1 = {}, acc2 = {}, acc3 = {};
#pragma unroll 4
    for (int k0 = 0; k0 < DIMX - 32; k0 += 32) {
        const int kn = k0 + 32;
        v16h a_n  = load_frag_a(Ab + kn, DIMX, lane);
        v16h b0_n = load_frag_b(W0 + kn, DIMX, lane);
        v16h b1_n = load_frag_b(W1 + kn, DIMX, lane);
        v16h b2_n = load_frag_b(W2 + kn, DIMX, lane);
        v16h b3_n = load_frag_b(W3 + kn, DIMX, lane);
        acc0 = WMMA_F16(a_c, b0_c, acc0);
        acc1 = WMMA_F16(a_c, b1_c, acc1);
        acc2 = WMMA_F16(a_c, b2_c, acc2);
        acc3 = WMMA_F16(a_c, b3_c, acc3);
        a_c = a_n; b0_c = b0_n; b1_c = b1_n; b2_c = b2_n; b3_c = b3_n;
    }
    acc0 = WMMA_F16(a_c, b0_c, acc0);
    acc1 = WMMA_F16(a_c, b1_c, acc1);
    acc2 = WMMA_F16(a_c, b2_c, acc2);
    acc3 = WMMA_F16(a_c, b3_c, acc3);

    const float qscale = 0.125f;   // HEAD_DIM^-0.5
    const int mrow = m0 + ((lane >> 4) << 3);
    v8f accs[4] = {acc0, acc1, acc2, acc3};
#pragma unroll
    for (int f = 0; f < 4; ++f) {
        const int col = n0 + f * 16 + (lane & 15);  // 0..1023 = h*64+d
        const int h = col >> 6, d = col & 63;
#pragma unroll
        for (int r = 0; r < 8; ++r) {
            const int m  = mrow + r;                 // 0..4095 = b*2048+n
            const int bb = m >> 11, n = m & (NSEQ - 1);
            const float v = accs[f][r];
            if (g < 3) {
                qb[((size_t)(bb * HEADS + h) * NSEQ + n) * KDIM + g * HD + d] =
                    (_Float16)(v * qscale);
            } else if (g < 6) {
                kb[((size_t)(bb * HEADS + h) * NSEQ + n) * KDIM + (g - 3) * HD + d] =
                    (_Float16)v;
            } else {
                vb[((size_t)(bb * HEADS + h) * HD + d) * NSEQ + n] = (_Float16)v;
            }
        }
    }
}

// ---------------------------------------------------------------------------
// Fused attention: one block = (b, h, 16 query rows), 8 waves, LDS-resident.
//  phase1: S[16][2048] = Q@K^T fp32 (n-tile pairs, K frags double-buffered)
//  phase2: row softmax (16 lanes/row, shuffle reduce), P = exp(S-max) f16
//  phase3: partial P@V^T per wave over a 256-key slice (pipelined)
//  phase4: LDS reduce over 8 waves, divide by row sum, store f16
// LDS: 128K (S) + 64K (P) + 32K (partials) + 64B = 229,440 B (< 320K/WGP)
// ---------------------------------------------------------------------------
__global__ __launch_bounds__(256) void attn_kernel(
    const _Float16* __restrict__ qb, const _Float16* __restrict__ kb,
    const _Float16* __restrict__ vb, _Float16* __restrict__ ao) {
    extern __shared__ char smem[];
    float*    S    = (float*)smem;                                   // [16][2048]
    _Float16* P    = (_Float16*)(smem + 16 * NSEQ * 4);              // [16][2048]
    float*    part = (float*)(smem + 16 * NSEQ * 4 + 16 * NSEQ * 2); // [8][16][64]
    float*    sums = (float*)(smem + 16 * NSEQ * 4 + 16 * NSEQ * 2 + 8 * 16 * 64 * 4);

    const int lane = threadIdx.x & 31;
    const int wave = threadIdx.x >> 5;
    const int q0 = (blockIdx.x & 127) << 4;
    const int h  = (blockIdx.x >> 7) & 15;
    const int b  = blockIdx.x >> 11;

    const _Float16* Q = qb + ((size_t)(b * HEADS + h) * NSEQ + q0) * KDIM;
    const _Float16* K = kb + ((size_t)(b * HEADS + h) * NSEQ) * KDIM;
    const _Float16* V = vb + ((size_t)(b * HEADS + h) * HD) * NSEQ;

    // ---- phase 1: scores (Q frags register-resident, shared by all tiles) --
    v16h af[6];
#pragma unroll
    for (int s = 0; s < 6; ++s) af[s] = load_frag_a(Q + s * 32, KDIM, lane);

#pragma unroll 1
    for (int pp = 0; pp < 8; ++pp) {
        const int nt0 = wave * 16 + pp * 2;
        const _Float16* Kt0 = K + (size_t)(nt0 * 16) * KDIM;
        const _Float16* Kt1 = Kt0 + (size_t)16 * KDIM;
        v8f acc0 = {}, acc1 = {};
        v16h b0_c = load_frag_b(Kt0, KDIM, lane);
        v16h b1_c = load_frag_b(Kt1, KDIM, lane);
#pragma unroll
        for (int s = 0; s < 5; ++s) {
            v16h b0_n = load_frag_b(Kt0 + (s + 1) * 32, KDIM, lane);
            v16h b1_n = load_frag_b(Kt1 + (s + 1) * 32, KDIM, lane);
            acc0 = WMMA_F16(af[s], b0_c, acc0);
            acc1 = WMMA_F16(af[s], b1_c, acc1);
            b0_c = b0_n; b1_c = b1_n;
        }
        acc0 = WMMA_F16(af[5], b0_c, acc0);
        acc1 = WMMA_F16(af[5], b1_c, acc1);

        const int col = nt0 * 16 + (lane & 15);
        const int row = (lane >> 4) << 3;
#pragma unroll
        for (int r = 0; r < 8; ++r) {
            S[(row + r) * NSEQ + col]      = acc0[r];
            S[(row + r) * NSEQ + col + 16] = acc1[r];
        }
    }
    __syncthreads();

    // ---- phase 2: softmax ------------------------------------------------
    {
        const int r = threadIdx.x >> 4;       // row 0..15
        const int j = threadIdx.x & 15;       // 16 lanes per row
        const float* rowp = S + (size_t)r * NSEQ + j * 128;
        float mx = -1e30f;
        for (int c = 0; c < 128; ++c) mx = fmaxf(mx, rowp[c]);
        for (int m = 8; m >= 1; m >>= 1) mx = fmaxf(mx, __shfl_xor(mx, m));
        float sm = 0.f;
        _Float16* prow = P + (size_t)r * NSEQ + j * 128;
        for (int c = 0; c < 128; ++c) {
            float e = __expf(rowp[c] - mx);
            sm += e;
            prow[c] = (_Float16)e;
        }
        for (int m = 8; m >= 1; m >>= 1) sm += __shfl_xor(sm, m);
        if (j == 0) sums[r] = sm;
    }
    __syncthreads();

    // ---- phase 3: P @ V^T (partial over this wave's 256-key slice) -------
    {
        const int kb0 = wave * 256;
        v16h a_c  = load_frag_a(P + kb0, NSEQ, lane);
        v16h b0_c = load_frag_b(V + (size_t) 0 * NSEQ + kb0, NSEQ, lane);
        v16h b1_c = load_frag_b(V + (size_t)16 * NSEQ + kb0, NSEQ, lane);
        v16h b2_c = load_frag_b(V + (size_t)32 * NSEQ + kb0, NSEQ, lane);
        v16h b3_c = load_frag_b(V + (size_t)48 * NSEQ + kb0, NSEQ, lane);
        v8f acc0 = {}, acc1 = {}, acc2 = {}, acc3 = {};
#pragma unroll
        for (int s = 0; s < 7; ++s) {
            const int kn = kb0 + (s + 1) * 32;
            v16h a_n  = load_frag_a(P + kn, NSEQ, lane);
            v16h b0_n = load_frag_b(V + (size_t) 0 * NSEQ + kn, NSEQ, lane);
            v16h b1_n = load_frag_b(V + (size_t)16 * NSEQ + kn, NSEQ, lane);
            v16h b2_n = load_frag_b(V + (size_t)32 * NSEQ + kn, NSEQ, lane);
            v16h b3_n = load_frag_b(V + (size_t)48 * NSEQ + kn, NSEQ, lane);
            acc0 = WMMA_F16(a_c, b0_c, acc0);
            acc1 = WMMA_F16(a_c, b1_c, acc1);
            acc2 = WMMA_F16(a_c, b2_c, acc2);
            acc3 = WMMA_F16(a_c, b3_c, acc3);
            a_c = a_n; b0_c = b0_n; b1_c = b1_n; b2_c = b2_n; b3_c = b3_n;
        }
        acc0 = WMMA_F16(a_c, b0_c, acc0);
        acc1 = WMMA_F16(a_c, b1_c, acc1);
        acc2 = WMMA_F16(a_c, b2_c, acc2);
        acc3 = WMMA_F16(a_c, b3_c, acc3);

        const int row = (lane >> 4) << 3;
        const int col = lane & 15;
        float* pw = part + (size_t)wave * (16 * 64);
        v8f pac[4] = {acc0, acc1, acc2, acc3};
#pragma unroll
        for (int f = 0; f < 4; ++f)
#pragma unroll
            for (int r = 0; r < 8; ++r)
                pw[(row + r) * 64 + f * 16 + col] = pac[f][r];
    }
    __syncthreads();

    // ---- phase 4: reduce, normalize, store -------------------------------
    {
        const int e0 = threadIdx.x * 4;
#pragma unroll
        for (int e = e0; e < e0 + 4; ++e) {
            const int r = e >> 6, d = e & 63;
            float s = 0.f;
            for (int w = 0; w < 8; ++w) s += part[w * (16 * 64) + e];
            const float val = s / sums[r];
            ao[((size_t)(b * NSEQ) + q0 + r) * DIMX + h * HD + d] = (_Float16)val;
        }
    }
}

// ---------------------------------------------------------------------------
// Output projection: OUT = attn_out @ w_out^T + b_out  (fp32 result)
// ---------------------------------------------------------------------------
__global__ __launch_bounds__(256) void outproj_kernel(
    const _Float16* __restrict__ A, const _Float16* __restrict__ W,
    const float* __restrict__ bias, float* __restrict__ out) {
    const int lane = threadIdx.x & 31;
    const int wave = threadIdx.x >> 5;
    const int t  = blockIdx.x * 8 + wave;
    const int m0 = (t >> 4) << 4;
    const int n0 = (t & 15) << 6;

    const _Float16* Ab = A + (size_t)m0 * DIMX;
    const _Float16* W0 = W + (size_t)(n0 +  0) * DIMX;
    const _Float16* W1 = W + (size_t)(n0 + 16) * DIMX;
    const _Float16* W2 = W + (size_t)(n0 + 32) * DIMX;
    const _Float16* W3 = W + (size_t)(n0 + 48) * DIMX;

    v16h a_c  = load_frag_a(Ab, DIMX, lane);
    v16h b0_c = load_frag_b(W0, DIMX, lane);
    v16h b1_c = load_frag_b(W1, DIMX, lane);
    v16h b2_c = load_frag_b(W2, DIMX, lane);
    v16h b3_c = load_frag_b(W3, DIMX, lane);

    v8f acc0 = {}, acc1 = {}, acc2 = {}, acc3 = {};
#pragma unroll 4
    for (int k0 = 0; k0 < DIMX - 32; k0 += 32) {
        const int kn = k0 + 32;
        v16h a_n  = load_frag_a(Ab + kn, DIMX, lane);
        v16h b0_n = load_frag_b(W0 + kn, DIMX, lane);
        v16h b1_n = load_frag_b(W1 + kn, DIMX, lane);
        v16h b2_n = load_frag_b(W2 + kn, DIMX, lane);
        v16h b3_n = load_frag_b(W3 + kn, DIMX, lane);
        acc0 = WMMA_F16(a_c, b0_c, acc0);
        acc1 = WMMA_F16(a_c, b1_c, acc1);
        acc2 = WMMA_F16(a_c, b2_c, acc2);
        acc3 = WMMA_F16(a_c, b3_c, acc3);
        a_c = a_n; b0_c = b0_n; b1_c = b1_n; b2_c = b2_n; b3_c = b3_n;
    }
    acc0 = WMMA_F16(a_c, b0_c, acc0);
    acc1 = WMMA_F16(a_c, b1_c, acc1);
    acc2 = WMMA_F16(a_c, b2_c, acc2);
    acc3 = WMMA_F16(a_c, b3_c, acc3);

    const int mrow = m0 + ((lane >> 4) << 3);
    v8f accs[4] = {acc0, acc1, acc2, acc3};
#pragma unroll
    for (int f = 0; f < 4; ++f) {
        const int col = n0 + f * 16 + (lane & 15);
        const float bv = bias[col];
#pragma unroll
        for (int r = 0; r < 8; ++r)
            out[(size_t)(mrow + r) * DIMX + col] = accs[f][r] + bv;
    }
}

// ---------------------------------------------------------------------------
// Host launcher
// ---------------------------------------------------------------------------
extern "C" void kernel_launch(void* const* d_in, const int* in_sizes, int n_in,
                              void* d_out, int out_size, void* d_ws, size_t ws_size,
                              hipStream_t stream) {
    (void)in_sizes; (void)n_in; (void)out_size; (void)ws_size;
    char* ws = (char*)d_ws;
    const size_t XE = (size_t)4096 * 1024;   // elements per [B,N,DIM] input
    const size_t WE = (size_t)1024 * 1024;   // elements per weight

    size_t o = 0;
    _Float16* xw    = (_Float16*)(ws + o); o += XE * 2;
    _Float16* xp    = (_Float16*)(ws + o); o += XE * 2;
    _Float16* xc    = (_Float16*)(ws + o); o += XE * 2;
    _Float16* wbase = (_Float16*)(ws + o); o += 8 * WE * 2;   // slots: wq,pq,cq,wk,pk,ck,wv,wout
    _Float16* qbuf  = (_Float16*)(ws + o); o += (size_t)NBAT * HEADS * NSEQ * KDIM * 2;
    _Float16* kbuf  = (_Float16*)(ws + o); o += (size_t)NBAT * HEADS * NSEQ * KDIM * 2;
    _Float16* vbuf  = (_Float16*)(ws + o); o += (size_t)NBAT * HEADS * HD * NSEQ * 2;
    _Float16* aout  = (_Float16*)(ws + o); o += XE * 2;

    // fp32 -> f16 conversions
    cvt_f32_f16<<<XE / 2048, 256, 0, stream>>>((const float*)d_in[0], xw, (int)XE);
    cvt_f32_f16<<<XE / 2048, 256, 0, stream>>>((const float*)d_in[1], xp, (int)XE);
    cvt_f32_f16<<<XE / 2048, 256, 0, stream>>>((const float*)d_in[2], xc, (int)XE);
    // weight slot order must match proj_kernel's g: {wq,pq,cq,wk,pk,ck,wv} + wout
    const int wmap[8] = {3, 6, 8, 4, 7, 9, 5, 10};
    for (int s = 0; s < 8; ++s)
        cvt_f32_f16<<<WE / 2048, 256, 0, stream>>>((const float*)d_in[wmap[s]],
                                                   wbase + (size_t)s * WE, (int)WE);

    // 7 projection GEMMs
    proj_kernel<<<dim3(512, 7), 256, 0, stream>>>(xw, xp, xc, wbase, qbuf, kbuf, vbuf);

    // fused attention: grid = B*H*(N/16) = 4096 blocks, 229,440 B LDS
    const size_t lds = 16 * NSEQ * 4 + 16 * NSEQ * 2 + 8 * 16 * 64 * 4 + 16 * 4;
    attn_kernel<<<NBAT * HEADS * (NSEQ / 16), 256, lds, stream>>>(qbuf, kbuf, vbuf, aout);

    // output projection + bias
    outproj_kernel<<<512, 256, 0, stream>>>(aout, wbase + (size_t)7 * WE,
                                            (const float*)d_in[11], (float*)d_out);
}
